// SelfAttention_63866163692092
// MI455X (gfx1250) — compile-verified, matrix operand order checked by reading, not run on hardware
//
#include <hip/hip_runtime.h>
#include <hip/hip_bf16.h>

// ---------------------------------------------------------------------------
// Self-attention for MI455X (gfx1250): bf16 WMMA (16x16x32) everywhere,
// fp32 accumulation, flash-attention so scores never touch HBM, and
// async-to-LDS (ASYNCcnt) double-buffered staging for the projection GEMMs.
// ---------------------------------------------------------------------------

typedef __bf16 bf16_t;
typedef __attribute__((ext_vector_type(16))) __bf16 v16bf;
typedef __attribute__((ext_vector_type(8)))  __bf16 v8bf;
typedef __attribute__((ext_vector_type(8)))  float  v8f;
typedef __attribute__((ext_vector_type(4)))  int    v4i;

#define B_  4
#define T_  2048
#define C_  1024
#define H_  16
#define HD_ 64

#define WMMA_BF16(a, b, c) \
    __builtin_amdgcn_wmma_f32_16x16x32_bf16(false, (a), false, (b), (short)0, (c), false, false)

#if defined(__has_builtin)
#if __has_builtin(__builtin_amdgcn_global_load_async_to_lds_b128) && \
    __has_builtin(__builtin_amdgcn_s_wait_asynccnt)
#define HAS_ASYNC_LDS 1
#endif
#endif
#ifndef HAS_ASYNC_LDS
#define HAS_ASYNC_LDS 0
#endif

// Builtin signature (from hipcc diagnostic): (v4i AS1*, v4i AS3*, imm, imm)
typedef __attribute__((address_space(1))) v4i gv4i_t;
typedef __attribute__((address_space(3))) v4i lv4i_t;

__device__ __forceinline__ v16bf make_frag(v8bf lo, v8bf hi) {
    union { v16bf v; v8bf h[2]; } u;
    u.h[0] = lo; u.h[1] = hi;
    return u.v;
}

// NT fragment load: works for both A (rows = M) and B (rows = N of W[N,K]).
// ISA 16-bit fragment layout: lane L holds row (L&15); its 16 bf16 values are
// two contiguous 16B chunks at k0 + 8*(L>=16) and that +16.
__device__ __forceinline__ v16bf load_frag_g(const bf16_t* __restrict__ base,
                                             int ld, int row0, int k0, int lane) {
    const int r  = row0 + (lane & 15);
    const int kk = k0 + ((lane >> 4) << 3);
    const bf16_t* p = base + (size_t)r * ld + kk;
    return make_frag(*(const v8bf*)p, *(const v8bf*)(p + 16));
}

__device__ __forceinline__ v16bf load_frag_lds(const bf16_t* p) {
    return make_frag(*(const v8bf*)p, *(const v8bf*)(p + 16));
}

// One 16-byte global -> LDS copy per lane. Async (ASYNCcnt) when available.
__device__ __forceinline__ void async_copy16(const bf16_t* __restrict__ g, bf16_t* l) {
#if HAS_ASYNC_LDS
    __builtin_amdgcn_global_load_async_to_lds_b128((gv4i_t*)g, (lv4i_t*)l, 0, 0);
#else
    *(v8bf*)l = *(const v8bf*)g;
#endif
}

// ---------------------------------------------------------------------------
// fp32 -> bf16 convert
// ---------------------------------------------------------------------------
__global__ void f32_to_bf16_kernel(const float* __restrict__ in,
                                   bf16_t* __restrict__ out, int n) {
    int idx = blockIdx.x * blockDim.x + threadIdx.x;
    if (idx < n) out[idx] = (bf16_t)in[idx];
}

// ---------------------------------------------------------------------------
// NT GEMM: out[M,N] = A[M,K] * Bm[N,K]^T + bias.  128x128 block tile,
// 8 waves (wave32) each computing 32x64 via 2x4 grid of 16x16 WMMA tiles.
// A/B k-chunks staged in LDS with double-buffered async copies: stage k+1 is
// in flight (ASYNCcnt) while WMMAs consume stage k from LDS.
// ---------------------------------------------------------------------------
#define LDS_STRIDE 40  // 32 + 8 bf16 pad: 80B rows -> conflict-free ds_load_b128

__global__ __launch_bounds__(256) void gemm_nt_wmma(
    const bf16_t* __restrict__ A,   // [M,K] row-major
    const bf16_t* __restrict__ Bm,  // [N,K] row-major (i.e. W, used as B^T)
    const float*  __restrict__ bias,// [N]
    bf16_t* __restrict__ outb,      // bf16 out (or null)
    float*  __restrict__ outf,      // f32 out (or null)
    int M, int N, int K) {
    __shared__ bf16_t sA[2][128 * LDS_STRIDE];
    __shared__ bf16_t sB[2][128 * LDS_STRIDE];

    const int tid  = threadIdx.x;
    const int lane = tid & 31;
    const int w    = tid >> 5;
    const int mblk = blockIdx.y * 128;
    const int nblk = blockIdx.x * 128;
    const int wm   = (w & 3) * 32;   // wave m offset inside tile
    const int wn   = (w >> 2) * 64;  // wave n offset inside tile

    // Stage one 128x32 A tile + 128x32 B tile: 512 16B chunks each,
    // thread handles chunks {tid, tid+256} -> exactly 4 async ops per thread.
    auto stage = [&](int buf, int k0) {
        for (int c = tid; c < 512; c += 256) {
            const int row = c >> 2;
            const int kp  = (c & 3) << 3;  // 0,8,16,24
            async_copy16(A  + (size_t)(mblk + row) * K + k0 + kp,
                         &sA[buf][row * LDS_STRIDE + kp]);
            async_copy16(Bm + (size_t)(nblk + row) * K + k0 + kp,
                         &sB[buf][row * LDS_STRIDE + kp]);
        }
    };

    v8f acc[2][4];
#pragma unroll
    for (int i = 0; i < 2; ++i)
#pragma unroll
        for (int j = 0; j < 4; ++j) acc[i][j] = (v8f)0.0f;

    const int frow = lane & 15;            // fragment row within 16
    const int fk   = (lane >> 4) << 3;     // +8 elements for upper half-wave

    stage(0, 0);
    int buf = 0;
    for (int k0 = 0; k0 < K; k0 += 32) {
        if (k0 + 32 < K) {
            stage(buf ^ 1, k0 + 32);       // prefetch next chunk (async)
#if HAS_ASYNC_LDS
            __builtin_amdgcn_s_wait_asynccnt(4);  // current chunk landed
#endif
        } else {
#if HAS_ASYNC_LDS
            __builtin_amdgcn_s_wait_asynccnt(0);
#endif
        }
        __syncthreads();

        v16bf a0 = load_frag_lds(&sA[buf][(wm + frow)      * LDS_STRIDE + fk]);
        v16bf a1 = load_frag_lds(&sA[buf][(wm + 16 + frow) * LDS_STRIDE + fk]);
        v16bf b0 = load_frag_lds(&sB[buf][(wn + frow)      * LDS_STRIDE + fk]);
        v16bf b1 = load_frag_lds(&sB[buf][(wn + 16 + frow) * LDS_STRIDE + fk]);
        v16bf b2 = load_frag_lds(&sB[buf][(wn + 32 + frow) * LDS_STRIDE + fk]);
        v16bf b3 = load_frag_lds(&sB[buf][(wn + 48 + frow) * LDS_STRIDE + fk]);
        acc[0][0] = WMMA_BF16(a0, b0, acc[0][0]);
        acc[0][1] = WMMA_BF16(a0, b1, acc[0][1]);
        acc[0][2] = WMMA_BF16(a0, b2, acc[0][2]);
        acc[0][3] = WMMA_BF16(a0, b3, acc[0][3]);
        acc[1][0] = WMMA_BF16(a1, b0, acc[1][0]);
        acc[1][1] = WMMA_BF16(a1, b1, acc[1][1]);
        acc[1][2] = WMMA_BF16(a1, b2, acc[1][2]);
        acc[1][3] = WMMA_BF16(a1, b3, acc[1][3]);

        __syncthreads();  // protect buffer reuse before next stage's writes
        buf ^= 1;
    }

    // C/D layout: VGPR j -> row m = j + 8*(lane>=16); column n = lane&15.
    const int moff = (lane >> 4) << 3;
    const int ncol = lane & 15;
#pragma unroll
    for (int mt = 0; mt < 2; ++mt) {
#pragma unroll
        for (int nt = 0; nt < 4; ++nt) {
            const int n = nblk + wn + nt * 16 + ncol;
            const float bz = bias[n];
#pragma unroll
            for (int j = 0; j < 8; ++j) {
                const int m = mblk + wm + mt * 16 + j + moff;
                const float val = acc[mt][nt][j] + bz;
                if (outb) outb[(size_t)m * N + n] = (bf16_t)val;
                else      outf[(size_t)m * N + n] = val;
            }
        }
    }
}

// ---------------------------------------------------------------------------
// RoPE + transpose: in [B,T,C] bf16 -> out [B,H,T,HD] bf16 (one thread/pair)
// ---------------------------------------------------------------------------
__global__ void rope_transpose_kernel(const bf16_t* __restrict__ in,
                                      const float* __restrict__ fc,  // [T,HD/2,2]
                                      bf16_t* __restrict__ out) {
    int idx = blockIdx.x * blockDim.x + threadIdx.x;
    const int NPAIR = B_ * H_ * T_ * (HD_ / 2);
    if (idx >= NPAIR) return;
    const int i   = idx % (HD_ / 2);
    int rem       = idx / (HD_ / 2);
    const int t   = rem % T_;
    const int bh  = rem / T_;
    const int b   = bh / H_;
    const int h   = bh % H_;
    const bf16_t* src = in + ((size_t)(b * T_ + t)) * C_ + h * HD_;
    const float x1 = (float)src[2 * i];
    const float x2 = (float)src[2 * i + 1];
    const float c  = fc[(t * (HD_ / 2) + i) * 2 + 0];
    const float s  = fc[(t * (HD_ / 2) + i) * 2 + 1];
    bf16_t* dst = out + ((size_t)bh * T_ + t) * HD_;
    dst[2 * i]     = (bf16_t)(x1 * c - x2 * s);
    dst[2 * i + 1] = (bf16_t)(x1 * s + x2 * c);
}

// V: [B,T,C] -> [B,H,HD,T] (d-major so PV B-fragments are contiguous over keys)
__global__ void transpose_v_kernel(const bf16_t* __restrict__ v0,
                                   bf16_t* __restrict__ vh) {
    int idx = blockIdx.x * blockDim.x + threadIdx.x;
    const int NE = B_ * H_ * HD_ * T_;
    if (idx >= NE) return;
    const int t  = idx % T_;
    int rem      = idx / T_;
    const int d  = rem % HD_;
    const int bh = rem / HD_;
    const int b  = bh / H_;
    const int h  = bh % H_;
    vh[idx] = v0[((size_t)(b * T_ + t)) * C_ + h * HD_ + d];
}

// ---------------------------------------------------------------------------
// Flash attention: grid (B*H, T/128), 256 threads (8 waves). Each wave owns
// 16 query rows; key tiles of 128; online softmax in fp32; P staged via LDS
// to re-layout into A-fragments for the PV WMMA.
// ---------------------------------------------------------------------------
__global__ __launch_bounds__(256) void flash_attn_wmma(
    const bf16_t* __restrict__ Q,   // [B,H,T,HD]
    const bf16_t* __restrict__ Kh,  // [B,H,T,HD]
    const bf16_t* __restrict__ Vt,  // [B,H,HD,T]
    const int*    __restrict__ pad, // [B,T]
    bf16_t* __restrict__ Y) {       // [B,T,C]
    __shared__ bf16_t p_lds[8 * 16 * 136];  // per-wave 16x128 P tile, padded

    const int lane  = threadIdx.x & 31;
    const int w     = threadIdx.x >> 5;
    const int bh    = blockIdx.x;
    const int b     = bh / H_;
    const int h     = bh % H_;
    const int qrow0 = blockIdx.y * 128 + w * 16;

    const bf16_t* qb = Q  + (size_t)bh * T_ * HD_;
    const bf16_t* kb = Kh + (size_t)bh * T_ * HD_;
    const bf16_t* vb = Vt + (size_t)bh * HD_ * T_;
    bf16_t* pl = &p_lds[w * 16 * 136];

    const int moff = (lane >> 4) << 3;  // rows j+8 for upper half-wave
    const int ncol = lane & 15;

    // Q fragments (HD=64 -> two K=32 steps), loaded once.
    const v16bf aq0 = load_frag_g(qb, HD_, qrow0, 0, lane);
    const v16bf aq1 = load_frag_g(qb, HD_, qrow0, 32, lane);

    v8f oacc[4];
    float mrow[8], lrow[8];
#pragma unroll
    for (int i = 0; i < 4; ++i) oacc[i] = (v8f)0.0f;
#pragma unroll
    for (int j = 0; j < 8; ++j) { mrow[j] = -1e30f; lrow[j] = 0.0f; }

    const int nktiles = blockIdx.y + 1;  // causal: keys up to this q block
    for (int kt = 0; kt < nktiles; ++kt) {
        const int kbase = kt * 128;

        // S = Q K^T : 8 n-tiles x 2 k-steps = 16 WMMA
        v8f s[8];
#pragma unroll
        for (int nt = 0; nt < 8; ++nt) s[nt] = (v8f)0.0f;
#pragma unroll
        for (int nt = 0; nt < 8; ++nt) {
            v16bf bk0 = load_frag_g(kb, HD_, kbase + nt * 16, 0, lane);
            v16bf bk1 = load_frag_g(kb, HD_, kbase + nt * 16, 32, lane);
            s[nt] = WMMA_BF16(aq0, bk0, s[nt]);
            s[nt] = WMMA_BF16(aq1, bk1, s[nt]);
        }

        // scale + causal/padding mask + running row max
        float nmax[8];
#pragma unroll
        for (int j = 0; j < 8; ++j) nmax[j] = mrow[j];
#pragma unroll
        for (int nt = 0; nt < 8; ++nt) {
            const int nk = kbase + nt * 16 + ncol;
            const bool ok = (pad[b * T_ + nk] != 0);
#pragma unroll
            for (int j = 0; j < 8; ++j) {
                const int mq = qrow0 + j + moff;
                float v = s[nt][j] * 0.125f;  // 1/sqrt(64)
                if (nk > mq || !ok) v = -1e30f;
                s[nt][j] = v;
                nmax[j] = fmaxf(nmax[j], v);
            }
        }
        // row values for row m live across one 16-lane half-group -> xor reduce
#pragma unroll
        for (int j = 0; j < 8; ++j) {
#pragma unroll
            for (int off = 1; off < 16; off <<= 1)
                nmax[j] = fmaxf(nmax[j], __shfl_xor(nmax[j], off, 32));
        }

        float alpha[8], psum[8];
#pragma unroll
        for (int j = 0; j < 8; ++j) {
            alpha[j] = __expf(mrow[j] - nmax[j]);
            mrow[j]  = nmax[j];
            psum[j]  = 0.0f;
        }

        // P = exp(S - max) -> LDS (row-major 16x128, stride 136)
#pragma unroll
        for (int nt = 0; nt < 8; ++nt) {
#pragma unroll
            for (int j = 0; j < 8; ++j) {
                const float p = __expf(s[nt][j] - nmax[j]);
                psum[j] += p;
                pl[(j + moff) * 136 + nt * 16 + ncol] = (bf16_t)p;
            }
        }
#pragma unroll
        for (int j = 0; j < 8; ++j) {
#pragma unroll
            for (int off = 1; off < 16; off <<= 1)
                psum[j] += __shfl_xor(psum[j], off, 32);
            lrow[j] = lrow[j] * alpha[j] + psum[j];
        }
#pragma unroll
        for (int nt4 = 0; nt4 < 4; ++nt4)
#pragma unroll
            for (int j = 0; j < 8; ++j) oacc[nt4][j] *= alpha[j];

        // O += P V : 4 k-steps x 4 HD n-tiles = 16 WMMA
#pragma unroll
        for (int ks = 0; ks < 4; ++ks) {
            const int r  = lane & 15;
            const int kk = ks * 32 + ((lane >> 4) << 3);
            const bf16_t* pp = &pl[r * 136 + kk];
            v16bf ap = make_frag(*(const v8bf*)pp, *(const v8bf*)(pp + 16));
#pragma unroll
            for (int nt4 = 0; nt4 < 4; ++nt4) {
                v16bf bvf = load_frag_g(vb, T_, nt4 * 16, kbase + ks * 32, lane);
                oacc[nt4] = WMMA_BF16(ap, bvf, oacc[nt4]);
            }
        }
    }

    // finalize: O /= l, write Y[B,T,C] bf16
#pragma unroll
    for (int j = 0; j < 8; ++j) {
        const float inv = (lrow[j] > 0.0f) ? (1.0f / lrow[j]) : 0.0f;
        const int t = qrow0 + j + moff;
#pragma unroll
        for (int nt4 = 0; nt4 < 4; ++nt4) {
            const int col = h * HD_ + nt4 * 16 + ncol;
            Y[((size_t)(b * T_ + t)) * C_ + col] = (bf16_t)(oacc[nt4][j] * inv);
        }
    }
}

// ---------------------------------------------------------------------------
extern "C" void kernel_launch(void* const* d_in, const int* in_sizes, int n_in,
                              void* d_out, int out_size, void* d_ws, size_t ws_size,
                              hipStream_t stream) {
    const float* x  = (const float*)d_in[0];
    const float* fc = (const float*)d_in[1];
    // d_in[2] = attn_mask (tril) -> causal computed analytically
    const int*   pad = (const int*)d_in[3];
    const float* Wq = (const float*)d_in[4];  const float* bq = (const float*)d_in[5];
    const float* Wk = (const float*)d_in[6];  const float* bk = (const float*)d_in[7];
    const float* Wv = (const float*)d_in[8];  const float* bv = (const float*)d_in[9];
    const float* Wp = (const float*)d_in[10]; const float* bp = (const float*)d_in[11];

    const size_t xN = (size_t)B_ * T_ * C_;  // 8388608
    const size_t wN = (size_t)C_ * C_;       // 1048576

    char* ws = (char*)d_ws;
    size_t off = 0;
    bf16_t* xb  = (bf16_t*)(ws + off); off += xN * 2;
    bf16_t* wqb = (bf16_t*)(ws + off); off += wN * 2;
    bf16_t* wkb = (bf16_t*)(ws + off); off += wN * 2;
    bf16_t* wvb = (bf16_t*)(ws + off); off += wN * 2;
    bf16_t* wpb = (bf16_t*)(ws + off); off += wN * 2;
    bf16_t* q0  = (bf16_t*)(ws + off); off += xN * 2;
    bf16_t* k0  = (bf16_t*)(ws + off); off += xN * 2;
    bf16_t* v0  = (bf16_t*)(ws + off); off += xN * 2;
    bf16_t* qh  = (bf16_t*)(ws + off); off += xN * 2;
    bf16_t* kh  = (bf16_t*)(ws + off); off += xN * 2;
    bf16_t* vh  = (bf16_t*)(ws + off); off += xN * 2;
    bf16_t* yh  = q0;  // q0 is dead after RoPE; reuse for attention output

    // 1) bf16 conversion
    f32_to_bf16_kernel<<<(int)((xN + 255) / 256), 256, 0, stream>>>(x, xb, (int)xN);
    f32_to_bf16_kernel<<<(int)((wN + 255) / 256), 256, 0, stream>>>(Wq, wqb, (int)wN);
    f32_to_bf16_kernel<<<(int)((wN + 255) / 256), 256, 0, stream>>>(Wk, wkb, (int)wN);
    f32_to_bf16_kernel<<<(int)((wN + 255) / 256), 256, 0, stream>>>(Wv, wvb, (int)wN);
    f32_to_bf16_kernel<<<(int)((wN + 255) / 256), 256, 0, stream>>>(Wp, wpb, (int)wN);

    // 2) QKV projections
    dim3 ggrid(C_ / 128, (B_ * T_) / 128);
    gemm_nt_wmma<<<ggrid, 256, 0, stream>>>(xb, wqb, bq, q0, nullptr, B_ * T_, C_, C_);
    gemm_nt_wmma<<<ggrid, 256, 0, stream>>>(xb, wkb, bk, k0, nullptr, B_ * T_, C_, C_);
    gemm_nt_wmma<<<ggrid, 256, 0, stream>>>(xb, wvb, bv, v0, nullptr, B_ * T_, C_, C_);

    // 3) RoPE + head transpose
    const int npairs = B_ * H_ * T_ * (HD_ / 2);
    rope_transpose_kernel<<<(npairs + 255) / 256, 256, 0, stream>>>(q0, fc, qh);
    rope_transpose_kernel<<<(npairs + 255) / 256, 256, 0, stream>>>(k0, fc, kh);
    const int nv = B_ * H_ * HD_ * T_;
    transpose_v_kernel<<<(nv + 255) / 256, 256, 0, stream>>>(v0, vh);

    // 4) flash attention
    flash_attn_wmma<<<dim3(B_ * H_, T_ / 128), 256, 0, stream>>>(qh, kh, vh, pad, yh);

    // 5) output projection -> fp32 d_out
    gemm_nt_wmma<<<ggrid, 256, 0, stream>>>(yh, wpb, bp, nullptr, (float*)d_out,
                                            B_ * T_, C_, C_);
}
